// ScaledDotProductAttention_78237124264536
// MI455X (gfx1250) — matrix operand order repaired and checked by reading, not compiled, not run on hardware
//
#include <hip/hip_runtime.h>

// Flash-attention forward, B=4,H=16,S=2048,D=64 fp32 -> gfx1250 f16 WMMA.
// v3: each wave owns 32 query rows (two 16-row subtiles) so the per-iteration
// shared costs (K/V f16 staging in LDS, barrier, K/V fragment ds_loads, loop
// SALU) amortize over 2x the matrix work: 18 wmma per 32-key step.
// Max-free softmax in exp2 domain (scale*log2e folded into Q); denominator
// via WMMA against an all-ones B tile; V staged transposed so B-fragments are
// two ds_load_b128 each.

typedef __attribute__((ext_vector_type(16))) _Float16 v16h;
typedef __attribute__((ext_vector_type(8)))  _Float16 v8h;
typedef __attribute__((ext_vector_type(8)))  float    v8f;

namespace {
constexpr int kB = 4, kH = 16, kS = 2048, kD = 64;
constexpr int kWaves = 8;            // 256 threads/block
constexpr int kKStr  = 72;           // K LDS row stride (halves), 144B (16B-aligned)
constexpr int kVStr  = 40;           // V^T LDS row stride (halves), 80B (16B-aligned)
constexpr int kPStr  = 40;           // P LDS row stride (halves)
constexpr float kQScale = 0.125f * 1.4426950408889634f;   // 1/sqrt(64) * log2(e)
constexpr float kMasked = -1.0e9f * 1.4426950408889634f;  // exp2 -> 0
}

__device__ __forceinline__ v8h cvt8(float4 a, float4 b) {
  v8h h;
  h[0]=(_Float16)a.x; h[1]=(_Float16)a.y; h[2]=(_Float16)a.z; h[3]=(_Float16)a.w;
  h[4]=(_Float16)b.x; h[5]=(_Float16)b.y; h[6]=(_Float16)b.z; h[7]=(_Float16)b.w;
  return h;
}

__global__ __launch_bounds__(kWaves*32) void fa_fwd(
    const float* __restrict__ Q, const float* __restrict__ K,
    const float* __restrict__ V, const unsigned char* __restrict__ Msk,
    float* __restrict__ Out)
{
  __shared__ __align__(16) _Float16 Ks[2][32*kKStr];      // [key][d]  f16
  __shared__ __align__(16) _Float16 Vs[2][64*kVStr];      // [dv][key] f16 (transposed)
  __shared__ __align__(16) _Float16 Ps[kWaves][32*kPStr]; // per-wave P (32 rows)

  const int tid  = threadIdx.x;
  const int lane = tid & 31;
  const int wave = tid >> 5;
  const int col  = lane & 15;   // N index inside a 16-wide tile
  const int hl   = lane >> 4;   // high half of the wave

  const int tile = blockIdx.x * kWaves + wave;   // [0, 4096): one 32-row Q tile
  const int qt = tile & (kS/32 - 1);             // 64 tiles per (b,h)
  const int bh = tile >> 6;
  const int b  = bh >> 4;

  const size_t base = (size_t)bh * kS * kD;
  const float* Kb = K + base;
  const float* Vb = V + base;
  const unsigned char* Mb = Msk + (size_t)b * (size_t)kS * kS;

  // cooperative staging role: 32 key-rows x 8 chunks of 8 elements
  const int srow = tid >> 3;
  const int sdc  = (tid & 7) * 8;

  // ---- Q: two 16-row subtiles, each as two f16 A-fragments, scale folded in.
  // 16-bit A 16x32 layout: lanes 0-15: K=0..7 (v0-3) / 16..23 (v4-7);
  // lanes 16-31: K=8..15 / 24..31.
  v16h qa[2][2];
  #pragma unroll
  for (int u = 0; u < 2; ++u) {
    const float* qp = Q + base + (size_t)(qt*32 + u*16 + col) * kD;
    #pragma unroll
    for (int f = 0; f < 2; ++f) {
      const float* p0 = qp + f*32 + hl*8;
      float4 a  = *(const float4*)(p0);
      float4 b4 = *(const float4*)(p0 + 4);
      float4 c  = *(const float4*)(p0 + 16);
      float4 d4 = *(const float4*)(p0 + 20);
      a.x*=kQScale; a.y*=kQScale; a.z*=kQScale; a.w*=kQScale;
      b4.x*=kQScale; b4.y*=kQScale; b4.z*=kQScale; b4.w*=kQScale;
      c.x*=kQScale; c.y*=kQScale; c.z*=kQScale; c.w*=kQScale;
      d4.x*=kQScale; d4.y*=kQScale; d4.z*=kQScale; d4.w*=kQScale;
      const v8h lo = cvt8(a, b4), hi8 = cvt8(c, d4);
      #pragma unroll
      for (int i = 0; i < 8; ++i) { qa[u][f][i] = lo[i]; qa[u][f][8+i] = hi8[i]; }
    }
  }

  // all-ones B fragment: o[u][4] accumulates the softmax denominator
  v16h onesf;
  #pragma unroll
  for (int i = 0; i < 16; ++i) onesf[i] = (_Float16)1.0f;

  // C/D layout: lanes 0-15 own rows 0..7, lanes 16-31 rows 8..15 of a subtile.
  const unsigned char* mbase[2];
  #pragma unroll
  for (int u = 0; u < 2; ++u)
    mbase[u] = Mb + (size_t)(qt*32 + u*16 + 8*hl) * kS;

  v8f o[2][5] = {};   // per subtile: 4 DV tiles + denominator

  for (int jt = 0; jt < kS/32; ++jt) {
    const int j = jt * 32;
    _Float16* ks = &Ks[jt & 1][0];
    _Float16* vs = &Vs[jt & 1][0];

    // mask bytes early so the loads overlap staging + barrier
    unsigned char mb0[2][8], mb1[2][8];
    #pragma unroll
    for (int u = 0; u < 2; ++u) {
      #pragma unroll
      for (int r = 0; r < 8; ++r) {
        mb0[u][r] = mbase[u][(size_t)r*kS + j + col];
        mb1[u][r] = mbase[u][(size_t)r*kS + j + 16 + col];
      }
    }

    // ---- cooperative f16 staging of the 32-key K and V tiles
    {
      const float* kp = Kb + (size_t)(j + srow)*kD + sdc;
      float4 a  = *(const float4*)(kp);
      float4 b4 = *(const float4*)(kp + 4);
      *(v8h*)(ks + srow*kKStr + sdc) = cvt8(a, b4);

      const float* vp = Vb + (size_t)(j + srow)*kD + sdc;
      float4 c  = *(const float4*)(vp);
      float4 d4 = *(const float4*)(vp + 4);
      const v8h vh = cvt8(c, d4);
      #pragma unroll
      for (int i = 0; i < 8; ++i) vs[(sdc + i)*kVStr + srow] = vh[i];  // transpose
    }
    __syncthreads();   // double-buffered: one barrier per iteration

    // ---- S = Q K^T: K fragments loaded once, used by both Q subtiles
    v8f s[2][2];
    #pragma unroll
    for (int u = 0; u < 2; ++u) { s[u][0] = (v8f){}; s[u][1] = (v8f){}; }
    #pragma unroll
    for (int t = 0; t < 2; ++t) {
      #pragma unroll
      for (int d = 0; d < 2; ++d) {
        const _Float16* kf = ks + (t*16 + col)*kKStr + d*32 + hl*8;
        const v8h lo = *(const v8h*)(kf), hi8 = *(const v8h*)(kf + 16);
        v16h kb;
        #pragma unroll
        for (int i = 0; i < 8; ++i) { kb[i] = lo[i]; kb[8+i] = hi8[i]; }
        s[0][t] = __builtin_amdgcn_wmma_f32_16x16x32_f16(
            false, qa[0][d], false, kb, (short)0, s[0][t], false, false);
        s[1][t] = __builtin_amdgcn_wmma_f32_16x16x32_f16(
            false, qa[1][d], false, kb, (short)0, s[1][t], false, false);
      }
    }

    // ---- max-free softmax numerator: p = exp2(x); mask -> 0 via underflow
    _Float16* pl = &Ps[wave][0];
    #pragma unroll
    for (int u = 0; u < 2; ++u) {
      #pragma unroll
      for (int r = 0; r < 8; ++r) {
        const float x0 = mb0[u][r] ? kMasked : s[u][0][r];
        const float x1 = mb1[u][r] ? kMasked : s[u][1][r];
        const float p0 = __builtin_amdgcn_exp2f(x0);
        const float p1 = __builtin_amdgcn_exp2f(x1);
        const int m = u*16 + r + 8*hl;
        pl[m*kPStr + col]      = (_Float16)p0;
        pl[m*kPStr + 16 + col] = (_Float16)p1;
      }
    }

    // ---- regather P as two 16x32 A-fragments (same-wave LDS transpose)
    v16h pa[2];
    #pragma unroll
    for (int u = 0; u < 2; ++u) {
      const v8h lo  = *(const v8h*)(pl + (u*16 + col)*kPStr + hl*8);
      const v8h hi8 = *(const v8h*)(pl + (u*16 + col)*kPStr + 16 + hl*8);
      #pragma unroll
      for (int i = 0; i < 8; ++i) { pa[u][i] = lo[i]; pa[u][8+i] = hi8[i]; }
    }

    // ---- context += P V ; each V fragment feeds both subtiles
    #pragma unroll
    for (int t = 0; t < 4; ++t) {
      const _Float16* vf = vs + (t*16 + col)*kVStr + hl*8;
      const v8h lo = *(const v8h*)(vf), hi8 = *(const v8h*)(vf + 16);
      v16h vb;
      #pragma unroll
      for (int i = 0; i < 8; ++i) { vb[i] = lo[i]; vb[8+i] = hi8[i]; }
      o[0][t] = __builtin_amdgcn_wmma_f32_16x16x32_f16(
          false, pa[0], false, vb, (short)0, o[0][t], false, false);
      o[1][t] = __builtin_amdgcn_wmma_f32_16x16x32_f16(
          false, pa[1], false, vb, (short)0, o[1][t], false, false);
    }
    o[0][4] = __builtin_amdgcn_wmma_f32_16x16x32_f16(
        false, pa[0], false, onesf, (short)0, o[0][4], false, false);
    o[1][4] = __builtin_amdgcn_wmma_f32_16x16x32_f16(
        false, pa[1], false, onesf, (short)0, o[1][4], false, false);
  }

  // ---- normalize by denominator and store (coalesced per half-wave)
  #pragma unroll
  for (int u = 0; u < 2; ++u) {
    #pragma unroll
    for (int r = 0; r < 8; ++r) {
      const float inv = 1.0f / o[u][4][r];
      float* op = Out + base + (size_t)(qt*32 + u*16 + r + 8*hl)*kD + col;
      op[0]  = o[u][0][r]*inv;
      op[16] = o[u][1][r]*inv;
      op[32] = o[u][2][r]*inv;
      op[48] = o[u][3][r]*inv;
    }
  }
}

extern "C" void kernel_launch(void* const* d_in, const int* in_sizes, int n_in,
                              void* d_out, int out_size, void* d_ws, size_t ws_size,
                              hipStream_t stream) {
  (void)in_sizes; (void)n_in; (void)out_size; (void)d_ws; (void)ws_size;
  const float* Q = (const float*)d_in[0];
  const float* K = (const float*)d_in[1];
  const float* V = (const float*)d_in[2];
  const unsigned char* M = (const unsigned char*)d_in[3];  // jax bool_ = 1 byte
  float* O = (float*)d_out;

  const int tiles = kB * kH * (kS / 32);   // 4096 wave tiles (32 rows each)
  dim3 grid(tiles / kWaves);               // 512 blocks x 256 threads
  fa_fwd<<<grid, kWaves * 32, 0, stream>>>(Q, K, V, M, O);
}
